// ReversedAttention_20555713479393
// MI455X (gfx1250) — compile-verified, matrix operand order checked by reading, not run on hardware
//
#include <hip/hip_runtime.h>
#include <math.h>
#include <stdint.h>

// ---------------------------------------------------------------------------
// ReversedAttention for MI455X (gfx1250, wave32, WMMA + async LDS DMA)
//   B=8, N=1024, C=768, H=12, D=64
// qkv / attention-out activations are stored f16 (WMMA operand precision),
// halving intermediate HBM traffic and removing downstream cvt VALU work.
// Attention is fused via  out[i] = sum_j exp(s_ji) * (p_j*m_j*g_j / Z_j) * v_j,
// Z_j = sum_i exp(s_ji): the post-softmax transpose + gating folds into one
// per-source coefficient, so the N x N matrix is never materialized.
// V tiles are staged into LDS with GLOBAL_LOAD_ASYNC_TO_LDS_B128 (ASYNCcnt).
// ---------------------------------------------------------------------------

#define BB 8
#define NN 1024
#define CC 768
#define HH 12
#define DD 64
#define QKVW (3 * CC) // 2304, row stride (in elements) of the qkv buffer

static __device__ __forceinline__ float ra_scale() { return 0.125f; } // D^-0.5

typedef __attribute__((ext_vector_type(16))) _Float16 v16h;
typedef __attribute__((ext_vector_type(8)))  _Float16 h8v;
typedef __attribute__((ext_vector_type(8)))  float    v8f;

// --- WMMA fragment loaders --------------------------------------------------
// 16-bit A fragment 16x32 (MxK): lane&15 = row, lane>>4 selects K half-group;
// each lane holds two contiguous 8-element chunks at k = kb+ksel and
// k = kb+ksel+16 (16B each).  B fragment is the mirrored layout (lane = N),
// so loading rows of a row-major [Nout,K] matrix with the same code gives B.

__device__ __forceinline__ v16h fragA_f32(const float* __restrict__ src, long ld,
                                          int rbase, int kbase, int lane) {
  const int ksel = (lane >> 4) << 3;
  const int r = lane & 15;
  const float* p = src + (long)(rbase + r) * ld + kbase + ksel;
  v16h f;
#pragma unroll
  for (int i = 0; i < 8; ++i) f[i] = (_Float16)p[i];
#pragma unroll
  for (int i = 0; i < 8; ++i) f[8 + i] = (_Float16)p[16 + i];
  return f;
}

__device__ __forceinline__ v16h fragA_h(const _Float16* __restrict__ src, long ld,
                                        int rbase, int kbase, int lane) {
  const int ksel = (lane >> 4) << 3;
  const int r = lane & 15;
  const _Float16* p = src + (long)(rbase + r) * ld + kbase + ksel; // 16B aligned
  const h8v lo = *(const h8v*)p;
  const h8v hi = *(const h8v*)(p + 16);
  v16h f;
#pragma unroll
  for (int i = 0; i < 8; ++i) { f[i] = lo[i]; f[8 + i] = hi[i]; }
  return f;
}

__device__ __forceinline__ v8f wmma_f16(v16h a, v16h b, v8f c) {
  return __builtin_amdgcn_wmma_f32_16x16x32_f16(false, a, false, b, (short)0, c,
                                                false, false);
}

// low 32 bits of a flat LDS pointer == LDS byte offset (ISA 10.2 aperture rule)
__device__ __forceinline__ unsigned lds_addr32(const void* p) {
  return (unsigned)(uintptr_t)p;
}

// ---------------------------------------------------------------------------
// Kernel 1/5: qkv projection, f16 output.  Y[8192,2304] = x @ qkv_w^T.
// Block = 4 waves, 64x64 block tile, 32x32 wave tile (2x2 WMMA).
// ---------------------------------------------------------------------------
__global__ __launch_bounds__(128) void ra_gemm_qkv(
    const float* __restrict__ A, const float* __restrict__ W,
    _Float16* __restrict__ Y) {
  const int lane = threadIdx.x & 31;
  const int wave = threadIdx.x >> 5;
  const int m0 = blockIdx.y * 64 + (wave >> 1) * 32;
  const int n0 = blockIdx.x * 64 + (wave & 1) * 32;
  const int r16 = lane & 15;

  v8f acc[2][2] = {};
  for (int kb = 0; kb < CC; kb += 32) {
    v16h a0 = fragA_f32(A, CC, m0, kb, lane);
    v16h a1 = fragA_f32(A, CC, m0 + 16, kb, lane);
    if (kb + 32 < CC)
      __builtin_prefetch(A + (long)(m0 + r16) * CC + kb + 32, 0, 1);
    v16h b0 = fragA_f32(W, CC, n0, kb, lane);
    v16h b1 = fragA_f32(W, CC, n0 + 16, kb, lane);
    acc[0][0] = wmma_f16(a0, b0, acc[0][0]);
    acc[0][1] = wmma_f16(a0, b1, acc[0][1]);
    acc[1][0] = wmma_f16(a1, b0, acc[1][0]);
    acc[1][1] = wmma_f16(a1, b1, acc[1][1]);
  }
  const int mh = (lane >> 4) << 3;
#pragma unroll
  for (int i = 0; i < 2; ++i)
#pragma unroll
    for (int j = 0; j < 2; ++j) {
      const int nb = n0 + j * 16 + r16;
#pragma unroll
      for (int r = 0; r < 8; ++r)
        Y[(long)(m0 + i * 16 + mh + r) * QKVW + nb] = (_Float16)acc[i][j][r];
    }
}

// ---------------------------------------------------------------------------
// Kernel 2/5: per-token head-selector softmax + message-controller gate.
// ---------------------------------------------------------------------------
__global__ __launch_bounds__(128) void ra_gate(
    const float* __restrict__ x, const float* __restrict__ m,
    const float* __restrict__ hs_w, const float* __restrict__ mc_w1,
    const float* __restrict__ mc_w2, float* __restrict__ coef,
    float* __restrict__ mout) {
  __shared__ float xs[CC];
  __shared__ float logits[HH];
  __shared__ float red[128];

  const long tok = blockIdx.x;
  const int b = (int)(tok >> 10), n = (int)(tok & 1023);
  const float* xr = x + tok * CC;
  for (int i = threadIdx.x; i < CC; i += 128) xs[i] = xr[i];
  __syncthreads();

  if (threadIdx.x < HH) {
    const float* w = hs_w + threadIdx.x * CC;
    float s = 0.f;
    for (int i = 0; i < CC; ++i) s = fmaf(xs[i], w[i], s);
    logits[threadIdx.x] = s;
  }

  float part = 0.f;
#pragma unroll
  for (int u = 0; u < 3; ++u) {
    const int j = threadIdx.x + u * 128;
    const float* w = mc_w1 + j * CC;
    float s = 0.f;
    for (int i = 0; i < CC; ++i) s = fmaf(xs[i], w[i], s);
    const float gact = 0.5f * s * (1.0f + erff(s * 0.70710678118654752f));
    part = fmaf(gact, mc_w2[j], part);
  }
  red[threadIdx.x] = part;
  __syncthreads();
#pragma unroll
  for (int off = 64; off > 0; off >>= 1) {
    if (threadIdx.x < off) red[threadIdx.x] += red[threadIdx.x + off];
    __syncthreads();
  }
  const float g = 1.0f / (1.0f + __expf(-red[0]));
  const float mo = m[tok] * g;
  if (threadIdx.x == 0) mout[tok] = mo;

  if (threadIdx.x < HH) {
    float mx = logits[0];
#pragma unroll
    for (int hh = 1; hh < HH; ++hh) mx = fmaxf(mx, logits[hh]);
    float den = 0.f;
#pragma unroll
    for (int hh = 0; hh < HH; ++hh) den += __expf(logits[hh] - mx);
    const float p = __expf(logits[threadIdx.x] - mx) / den;
    coef[((long)(b * HH + threadIdx.x)) * NN + n] = p * mo;
  }
}

// ---------------------------------------------------------------------------
// Kernel 3/5: Z[b,h,j] = sum_i exp(scale * q_j . k_i).
// Wave owns 64 Q rows (8 resident A frags), streams K rows as B frags.
// ---------------------------------------------------------------------------
__global__ __launch_bounds__(128) void ra_zsum(const _Float16* __restrict__ qkv,
                                               float* __restrict__ Z) {
  const int lane = threadIdx.x & 31;
  const int gid = blockIdx.x * 4 + (threadIdx.x >> 5); // 0..1535
  const int jb = (gid & 15) * 64;
  const int bh = gid >> 4;
  const int b = bh / HH, h = bh - b * HH;
  const _Float16* q = qkv + (long)b * NN * QKVW + h * DD;
  const _Float16* kmat = q + CC;

  v16h qa[4][2];
#pragma unroll
  for (int t = 0; t < 4; ++t) {
    qa[t][0] = fragA_h(q, QKVW, jb + t * 16, 0, lane);
    qa[t][1] = fragA_h(q, QKVW, jb + t * 16, 32, lane);
  }

  v8f z[4] = {};
  for (int i = 0; i < NN; i += 16) {
    v16h kb0 = fragA_h(kmat, QKVW, i, 0, lane);
    v16h kb1 = fragA_h(kmat, QKVW, i, 32, lane);
    if (i + 16 < NN)
      __builtin_prefetch(kmat + (long)(i + 16 + (lane & 15)) * QKVW, 0, 1);
#pragma unroll
    for (int t = 0; t < 4; ++t) {
      v8f s = {};
      s = wmma_f16(qa[t][0], kb0, s);
      s = wmma_f16(qa[t][1], kb1, s);
#pragma unroll
      for (int r = 0; r < 8; ++r) z[t][r] += __expf(s[r] * ra_scale());
    }
  }
#pragma unroll
  for (int t = 0; t < 4; ++t)
#pragma unroll
    for (int r = 0; r < 8; ++r) {
      float v = z[t][r];
      v += __shfl_xor(v, 1);
      v += __shfl_xor(v, 2);
      v += __shfl_xor(v, 4);
      v += __shfl_xor(v, 8);
      if ((lane & 15) == 0)
        Z[(long)bh * NN + jb + t * 16 + ((lane >> 4) << 3) + r] = v;
    }
}

// ---------------------------------------------------------------------------
// Kernel 4/5: fused reversed attention output (f16 in, f16 out).
//   O[b,h,i,:] = sum_j exp(scale * k_i . q_j) * (coef_j / Z_j) * v[b,h,j,:]
// Per 32-wide j step: V tile (32x64 f16 = 4KB) is DMA'd into LDS with
// GLOBAL_LOAD_ASYNC_TO_LDS_B128 (no VGPR round trip, ASYNCcnt tracked),
// score tiles via WMMA, exp + per-lane column coefficient, P restaged as an
// f16 A fragment through LDS, V B-frags read from LDS (strided = transpose).
// ---------------------------------------------------------------------------
__global__ __launch_bounds__(128) void ra_attn_out(
    const _Float16* __restrict__ qkv, const float* __restrict__ coef,
    const float* __restrict__ Z, _Float16* __restrict__ Obuf) {
  __shared__ _Float16 Vs[32][DD];     // raw row-major V tile (async DMA target)
  __shared__ _Float16 Ps[4][16][32];  // per-wave P restage (16 i x 32 j)

  const int lane = threadIdx.x & 31;
  const int wave = threadIdx.x >> 5;
  const int bh = blockIdx.x >> 4;
  const int iblk = blockIdx.x & 15;
  const int b = bh / HH, h = bh - b * HH;
  const int it = iblk * 64 + wave * 16;

  const _Float16* q = qkv + (long)b * NN * QKVW + h * DD;
  const _Float16* kmat = q + CC;
  const _Float16* vmat = q + 2 * CC;
  const float* cz_c = coef + (long)bh * NN;
  const float* cz_z = Z + (long)bh * NN;

  const v16h ka0 = fragA_h(kmat, QKVW, it, 0, lane);
  const v16h ka1 = fragA_h(kmat, QKVW, it, 32, lane);

  const int r16 = lane & 15;
  const int mh = (lane >> 4) << 3;
  const unsigned vs_base = lds_addr32(&Vs[0][0]);

  v8f oacc[4] = {};

  for (int j = 0; j < NN; j += 32) {
    __syncthreads(); // previous Vs fully consumed by all waves

    // async DMA: 256 x 16B chunks; each thread issues 2 B128 copies
#pragma unroll
    for (int u = 0; u < 2; ++u) {
      const int c = threadIdx.x + u * 128;
      const int jj = c >> 3;          // tile row
      const int o8 = (c & 7) * 8;     // halfword offset in row
      const uint64_t ga = (uint64_t)(uintptr_t)(vmat + (long)(j + jj) * QKVW + o8);
      const unsigned la = vs_base + (unsigned)((jj * DD + o8) * 2);
      asm volatile("global_load_async_to_lds_b128 %0, %1, off"
                   :: "v"(la), "v"(ga) : "memory");
    }
    asm volatile("s_wait_asynccnt 0x0" ::: "memory");
    __syncthreads(); // everyone's DMA landed

    // score tiles S'[i, j..j+31] = K_i . Q_j
    v16h qb0 = fragA_h(q, QKVW, j, 0, lane);
    v16h qb1 = fragA_h(q, QKVW, j, 32, lane);
    v16h qb2 = fragA_h(q, QKVW, j + 16, 0, lane);
    v16h qb3 = fragA_h(q, QKVW, j + 16, 32, lane);
    v8f s0 = {}, s1 = {};
    s0 = wmma_f16(ka0, qb0, s0);
    s0 = wmma_f16(ka1, qb1, s0);
    s1 = wmma_f16(ka0, qb2, s1);
    s1 = wmma_f16(ka1, qb3, s1);

    const float c0 = cz_c[j + r16] / cz_z[j + r16];
    const float c1 = cz_c[j + 16 + r16] / cz_z[j + 16 + r16];

#pragma unroll
    for (int r = 0; r < 8; ++r) {
      Ps[wave][mh + r][r16]      = (_Float16)(__expf(s0[r] * ra_scale()) * c0);
      Ps[wave][mh + r][16 + r16] = (_Float16)(__expf(s1[r] * ra_scale()) * c1);
    }

    const v16h pa = fragA_h(&Ps[wave][0][0], 32, 0, 0, lane);
#pragma unroll
    for (int dt = 0; dt < 4; ++dt) {
      // B[k=jj][n=d] via strided (transposing) LDS reads
      const int d = dt * 16 + r16;
      const int ksel = mh; // (lane>>4)*8
      v16h vf;
#pragma unroll
      for (int i = 0; i < 8; ++i) {
        vf[i]     = Vs[ksel + i][d];
        vf[8 + i] = Vs[16 + ksel + i][d];
      }
      oacc[dt] = wmma_f16(pa, vf, oacc[dt]);
    }
  }

#pragma unroll
  for (int dt = 0; dt < 4; ++dt)
#pragma unroll
    for (int r = 0; r < 8; ++r)
      Obuf[((long)bh * NN + it + mh + r) * DD + dt * 16 + r16] =
          (_Float16)oacc[dt][r];
}

// ---------------------------------------------------------------------------
// Kernel 5/5: head concat + output projection.  out = O @ proj_w^T + proj_b.
// A rows gathered from f16 [B,H,N,D] with k = h*64+d; W/bias/out are f32.
// ---------------------------------------------------------------------------
__global__ __launch_bounds__(128) void ra_gemm_proj(
    const _Float16* __restrict__ Af, const float* __restrict__ W,
    const float* __restrict__ bias, float* __restrict__ Y) {
  const int lane = threadIdx.x & 31;
  const int wave = threadIdx.x >> 5;
  const int m0 = blockIdx.y * 64 + (wave >> 1) * 32;
  const int n0 = blockIdx.x * 64 + (wave & 1) * 32;
  const int ksel = (lane >> 4) << 3;
  const int r16 = lane & 15;

  v8f acc[2][2] = {};
  for (int kb = 0; kb < CC; kb += 32) {
    const int h = kb >> 6;               // 32-chunk stays inside one head
    const int d = (kb & 63) + ksel;
    v16h a0, a1;
#pragma unroll
    for (int t = 0; t < 2; ++t) {
      const int row = m0 + t * 16 + r16;
      const int b = row >> 10, n = row & 1023;
      const _Float16* p = Af + ((long)(b * HH + h) * NN + n) * DD + d;
      const h8v lo = *(const h8v*)p;
      const h8v hi = *(const h8v*)(p + 16);
      v16h f;
#pragma unroll
      for (int i = 0; i < 8; ++i) { f[i] = lo[i]; f[8 + i] = hi[i]; }
      if (t == 0) a0 = f; else a1 = f;
    }
    v16h b0 = fragA_f32(W, CC, n0, kb, lane);
    v16h b1 = fragA_f32(W, CC, n0 + 16, kb, lane);
    acc[0][0] = wmma_f16(a0, b0, acc[0][0]);
    acc[0][1] = wmma_f16(a0, b1, acc[0][1]);
    acc[1][0] = wmma_f16(a1, b0, acc[1][0]);
    acc[1][1] = wmma_f16(a1, b1, acc[1][1]);
  }

  const int mh = (lane >> 4) << 3;
#pragma unroll
  for (int i = 0; i < 2; ++i)
#pragma unroll
    for (int j = 0; j < 2; ++j) {
      const int nb = n0 + j * 16 + r16;
      const float bi = bias[nb];
#pragma unroll
      for (int r = 0; r < 8; ++r)
        Y[(long)(m0 + i * 16 + mh + r) * CC + nb] = acc[i][j][r] + bi;
    }
}

// ---------------------------------------------------------------------------
// Launch
// ---------------------------------------------------------------------------
extern "C" void kernel_launch(void* const* d_in, const int* in_sizes, int n_in,
                              void* d_out, int out_size, void* d_ws,
                              size_t ws_size, hipStream_t stream) {
  const float* x      = (const float*)d_in[0];
  const float* m      = (const float*)d_in[1];
  const float* qkv_w  = (const float*)d_in[2];
  const float* hs_w   = (const float*)d_in[3];
  const float* mc_w1  = (const float*)d_in[4];
  const float* mc_w2  = (const float*)d_in[5];
  const float* proj_w = (const float*)d_in[6];
  const float* proj_b = (const float*)d_in[7];

  float* out  = (float*)d_out;
  float* mout = out + (size_t)BB * NN * CC; // tuple part 2: m_out [B,1,1,N]

  _Float16* qkv_h  = (_Float16*)d_ws;                       // 18,874,368 h (36MB)
  _Float16* obuf_h = qkv_h + (size_t)BB * NN * QKVW;        //  6,291,456 h (12MB)
  float*    coef   = (float*)(obuf_h + (size_t)BB * HH * NN * DD); // [B,H,N]
  float*    Zb     = coef + (size_t)BB * HH * NN;                  // [B,H,N]

  // 1) qkv projection (f16 out)
  ra_gemm_qkv<<<dim3(QKVW / 64, (BB * NN) / 64), 128, 0, stream>>>(x, qkv_w,
                                                                   qkv_h);
  // 2) head-selector softmax * message gate -> coef, m_out
  ra_gate<<<BB * NN, 128, 0, stream>>>(x, m, hs_w, mc_w1, mc_w2, coef, mout);

  // 3) softmax denominators Z_j
  ra_zsum<<<(BB * HH * (NN / 64)) / 4, 128, 0, stream>>>(qkv_h, Zb);

  // 4) fused reversed-attention output (async LDS DMA for V tiles)
  ra_attn_out<<<BB * HH * (NN / 64), 128, 0, stream>>>(qkv_h, coef, Zb, obuf_h);

  // 5) head concat + output projection (+bias)
  ra_gemm_proj<<<dim3(CC / 64, (BB * NN) / 64), 128, 0, stream>>>(
      obuf_h, proj_w, proj_b, out);
}
